// MultiScaleGATEncoder_44401371906309
// MI455X (gfx1250) — matrix operand order repaired and checked by reading, not compiled, not run on hardware
//
#include <hip/hip_runtime.h>
#include <hip/hip_bf16.h>

// ---------------------------------------------------------------------------
// MultiScaleGATEncoder for MI455X (gfx1250, wave32).
// Dense GEMMs -> v_wmma_f32_16x16x32_f16 fed by fragment-ready LDS panels
// (2x ds_load_b128 per fragment). Irregular segment softmax / scatter
// aggregation -> VALU + global atomics.
// ---------------------------------------------------------------------------

typedef __attribute__((ext_vector_type(16))) _Float16 v16h;
typedef __attribute__((ext_vector_type(8)))  float    v8f;

#define NNODES 6000

union FragU { v16h h; float4 f4[2]; };           // 32B WMMA operand
union PackU { float4 f4; _Float16 h[8]; };       // 16B half-chunk builder

// ---------------------------------------------------------------------------
// WMMA GEMM:  C[M,N] = act(A[M,K] @ B[K,N] + bias),  K multiple of 32, K<=256.
// Block = 128 threads (4 waves); block tile 64x64; wave tile 16x64.
// Whole A(64xK) and B(Kx64) panels are staged once into dynamic LDS in
// fragment-ready layout; the K-loop is then pure ds_load_b128 + v_wmma.
//   A frag (lane l, wave w, kstep ks): halves e -> A[m0+w*16+(l&15)]
//        [ks*32 + (l>>4)*8 + (e<8 ? e : 8+e)]   == 2 contiguous 8-half runs
//   B frag (lane l, jtile j, kstep ks): halves e -> B[ks*32+(l>>4)*8+(e<8 ? e : 8+e)]
//        [n0 + j*16 + (l&15)]
// LDS layout: panel[((ks*4 + wj)*2 + chunk)*32 + lane] as float4 (16B).
// act: 0=none, 1=relu, 2=tanh
// ---------------------------------------------------------------------------
__global__ __launch_bounds__(128) void wmma_gemm_kernel(
    const float* __restrict__ A, const float* __restrict__ B,
    const float* __restrict__ bias, float* __restrict__ C,
    int M, int K, int N, int act)
{
    extern __shared__ float4 smem4[];
    const int KS = K >> 5;                         // K-steps of 32
    float4* AsF = smem4;                           // KS*4*2*32 float4
    float4* BsF = smem4 + (size_t)KS * 256;        // KS*4*2*32 float4

    const int tid  = threadIdx.x;
    const int wave = tid >> 5;                     // 0..3 (= wj for writers)
    const int lane = tid & 31;
    const int lm   = lane & 15;
    const int lhi  = lane >> 4;
    const int m0   = blockIdx.x * 64;
    const int n0   = blockIdx.y * 64;

    // ---- stage A and B panels into fragment-ready LDS ---------------------
    {
        const int gm = m0 + wave * 16 + lm;        // A row this thread owns
        const int gn = n0 + wave * 16 + lm;        // B col this thread owns
        for (int ks = 0; ks < KS; ++ks) {
            // A: two contiguous 8-half runs per (lane,ks)
            PackU a0, a1;
            const int kb = ks * 32 + lhi * 8;
            if (gm < M) {
                const float* ap = A + (size_t)gm * K + kb;
                float4 v0 = *(const float4*)(ap);
                float4 v1 = *(const float4*)(ap + 4);
                float4 v2 = *(const float4*)(ap + 16);
                float4 v3 = *(const float4*)(ap + 20);
                a0.h[0] = (_Float16)v0.x; a0.h[1] = (_Float16)v0.y;
                a0.h[2] = (_Float16)v0.z; a0.h[3] = (_Float16)v0.w;
                a0.h[4] = (_Float16)v1.x; a0.h[5] = (_Float16)v1.y;
                a0.h[6] = (_Float16)v1.z; a0.h[7] = (_Float16)v1.w;
                a1.h[0] = (_Float16)v2.x; a1.h[1] = (_Float16)v2.y;
                a1.h[2] = (_Float16)v2.z; a1.h[3] = (_Float16)v2.w;
                a1.h[4] = (_Float16)v3.x; a1.h[5] = (_Float16)v3.y;
                a1.h[6] = (_Float16)v3.z; a1.h[7] = (_Float16)v3.w;
            } else {
                for (int e = 0; e < 8; ++e) { a0.h[e] = (_Float16)0.0f; a1.h[e] = (_Float16)0.0f; }
            }
            AsF[((ks * 4 + wave) * 2 + 0) * 32 + lane] = a0.f4;
            AsF[((ks * 4 + wave) * 2 + 1) * 32 + lane] = a1.f4;

            // B: two 8-k runs of one column, same k order as A (coalesced across lanes)
            PackU b0, b1;
            const int kbb = ks * 32 + lhi * 8;
            if (gn < N) {
                const float* bp = B + (size_t)kbb * N + gn;
                for (int e = 0; e < 8; ++e)
                    b0.h[e] = (_Float16)bp[(size_t)e * N];
                for (int e = 0; e < 8; ++e)
                    b1.h[e] = (_Float16)bp[(size_t)(16 + e) * N];
            } else {
                for (int e = 0; e < 8; ++e) { b0.h[e] = (_Float16)0.0f; b1.h[e] = (_Float16)0.0f; }
            }
            BsF[((ks * 4 + wave) * 2 + 0) * 32 + lane] = b0.f4;
            BsF[((ks * 4 + wave) * 2 + 1) * 32 + lane] = b1.f4;
        }
    }
    __syncthreads();

    // ---- barrier-free compute loop: ds_load_b128 + v_wmma -----------------
    v8f acc[4];
    for (int j = 0; j < 4; ++j)
        for (int r = 0; r < 8; ++r) acc[j][r] = 0.0f;

    for (int ks = 0; ks < KS; ++ks) {
        FragU a;
        const float4* ab = &AsF[((ks * 4 + wave) * 2) * 32 + lane];
        a.f4[0] = ab[0];
        a.f4[1] = ab[32];
        for (int j = 0; j < 4; ++j) {
            FragU b;
            const float4* bb = &BsF[((ks * 4 + j) * 2) * 32 + lane];
            b.f4[0] = bb[0];
            b.f4[1] = bb[32];
            acc[j] = __builtin_amdgcn_wmma_f32_16x16x32_f16(
                false, a.h, false, b.h, (short)0, acc[j], false, false);
        }
    }

    // ---- epilogue: bias + activation + guarded store ----------------------
    for (int j = 0; j < 4; ++j) {
        int gn = n0 + j * 16 + lm;
        if (gn >= N) continue;
        float bv = bias ? bias[gn] : 0.0f;
        for (int r = 0; r < 8; ++r) {
            int gm = m0 + wave * 16 + lhi * 8 + r;
            if (gm >= M) continue;
            float v = acc[j][r] + bv;
            if (act == 1)      v = fmaxf(v, 0.0f);
            else if (act == 2) v = tanhf(v);
            C[(size_t)gm * N + gn] = v;
        }
    }
}

// ---------------------------------------------------------------------------
// Fill helpers (graph-capture-safe replacement for hipMemsetAsync)
// ---------------------------------------------------------------------------
__global__ void fill_f32_kernel(float* p, float v, int n) {
    int i = blockIdx.x * blockDim.x + threadIdx.x;
    if (i < n) p[i] = v;
}
__global__ void fill_u32_kernel(unsigned* p, unsigned v, int n) {
    int i = blockIdx.x * blockDim.x + threadIdx.x;
    if (i < n) p[i] = v;
}

// ---------------------------------------------------------------------------
// a_src[n,h] = <h[n,h,:], att_src[h,:]>, likewise a_dst. One wave per node.
// ---------------------------------------------------------------------------
__global__ __launch_bounds__(256) void attn_scores_kernel(
    const float* __restrict__ h,
    const float* __restrict__ att_src, const float* __restrict__ att_dst,
    float* __restrict__ asrc, float* __restrict__ adst, int n)
{
    int w    = (blockIdx.x * blockDim.x + threadIdx.x) >> 5;
    int lane = threadIdx.x & 31;
    if (w >= n) return;
    const float* hr = h + (size_t)w * 256;
    for (int hd = 0; hd < 4; ++hd) {
        float x0 = hr[hd*64 + lane], x1 = hr[hd*64 + 32 + lane];
        float s = x0 * att_src[hd*64 + lane] + x1 * att_src[hd*64 + 32 + lane];
        float d = x0 * att_dst[hd*64 + lane] + x1 * att_dst[hd*64 + 32 + lane];
        for (int off = 16; off > 0; off >>= 1) {
            s += __shfl_down(s, off, 32);
            d += __shfl_down(d, off, 32);
        }
        if (lane == 0) { asrc[w*4 + hd] = s; adst[w*4 + hd] = d; }
    }
}

// float <-> order-preserving unsigned, for atomicMax-based segment max
__device__ __forceinline__ unsigned fenc(float f) {
    unsigned u = __float_as_uint(f);
    return (u & 0x80000000u) ? ~u : (u | 0x80000000u);
}
__device__ __forceinline__ float fdec(unsigned u) {
    return __uint_as_float((u & 0x80000000u) ? (u & 0x7fffffffu) : ~u);
}

__device__ __forceinline__ void edge_sd(int e, int E, const int* src, const int* dst,
                                        int& s, int& d) {
    if (e < E) { s = src[e]; d = dst[e]; }
    else       { s = e - E;  d = e - E;  }   // appended self loops
}

// thread per (edge, head): segment max over destination
__global__ void edge_max_kernel(const int* __restrict__ src, const int* __restrict__ dst,
                                int E, int n, const float* __restrict__ asrc,
                                const float* __restrict__ adst, unsigned* __restrict__ emax)
{
    int idx = blockIdx.x * blockDim.x + threadIdx.x;
    if (idx >= (E + n) * 4) return;
    int e = idx >> 2, hd = idx & 3, s, d;
    edge_sd(e, E, src, dst, s, d);
    float v = asrc[s*4 + hd] + adst[d*4 + hd];
    v = (v > 0.0f) ? v : 0.2f * v;                 // leaky relu
    atomicMax(&emax[d*4 + hd], fenc(v));
}

// thread per (edge, head): exp(e - max) and segment-sum denominator
__global__ void edge_exp_kernel(const int* __restrict__ src, const int* __restrict__ dst,
                                int E, int n, const float* __restrict__ asrc,
                                const float* __restrict__ adst,
                                const unsigned* __restrict__ emax,
                                float* __restrict__ eexp, float* __restrict__ denom)
{
    int idx = blockIdx.x * blockDim.x + threadIdx.x;
    if (idx >= (E + n) * 4) return;
    int e = idx >> 2, hd = idx & 3, s, d;
    edge_sd(e, E, src, dst, s, d);
    float v = asrc[s*4 + hd] + adst[d*4 + hd];
    v = (v > 0.0f) ? v : 0.2f * v;
    float ex = __expf(v - fdec(emax[d*4 + hd]));
    eexp[idx] = ex;
    atomicAdd(&denom[d*4 + hd], ex);
}

// thread per (edge, head-channel): agg[dst] += alpha * h[src]
__global__ void aggregate_kernel(const int* __restrict__ src, const int* __restrict__ dst,
                                 int E, int n, const float* __restrict__ h,
                                 const float* __restrict__ eexp,
                                 const float* __restrict__ denom, float* __restrict__ agg)
{
    long long idx = (long long)blockIdx.x * blockDim.x + threadIdx.x;
    if (idx >= (long long)(E + n) * 256) return;
    int e = (int)(idx >> 8), hc = (int)(idx & 255), hd = hc >> 6, s, d;
    edge_sd(e, E, src, dst, s, d);
    float alpha = eexp[e*4 + hd] / (denom[d*4 + hd] + 1e-16f);
    atomicAdd(&agg[(size_t)d * 256 + hc], h[(size_t)s * 256 + hc] * alpha);
}

// heads concat + bias   /   heads mean + bias
__global__ void bias_concat_kernel(const float* __restrict__ agg,
                                   const float* __restrict__ bias,
                                   float* __restrict__ out, int n)
{
    int idx = blockIdx.x * blockDim.x + threadIdx.x;
    if (idx >= n * 256) return;
    out[idx] = agg[idx] + bias[idx & 255];
}
__global__ void bias_mean_kernel(const float* __restrict__ agg,
                                 const float* __restrict__ bias,
                                 float* __restrict__ out, int n)
{
    int idx = blockIdx.x * blockDim.x + threadIdx.x;
    if (idx >= n * 64) return;
    int c = idx & 63, r = idx >> 6;
    const float* a = agg + (size_t)r * 256;
    out[idx] = 0.25f * (a[c] + a[64 + c] + a[128 + c] + a[192 + c]) + bias[c];
}

// LayerNorm + sigmoid(rw)-gated residual + optional ELU.  One wave per row.
__global__ __launch_bounds__(256) void ln_res_kernel(
    const float* __restrict__ xg, const float* __restrict__ g,
    const float* __restrict__ b,  const float* __restrict__ proj,
    const float* __restrict__ rw, float* __restrict__ out,
    int n, int D, int do_elu)
{
    int row  = (blockIdx.x * blockDim.x + threadIdx.x) >> 5;
    int lane = threadIdx.x & 31;
    if (row >= n) return;
    const float* xr = xg + (size_t)row * D;
    float s = 0.0f;
    for (int c = lane; c < D; c += 32) s += xr[c];
    for (int off = 16; off > 0; off >>= 1) s += __shfl_down(s, off, 32);
    float mu = __shfl(s, 0, 32) / (float)D;
    float v = 0.0f;
    for (int c = lane; c < D; c += 32) { float t = xr[c] - mu; v += t * t; }
    for (int off = 16; off > 0; off >>= 1) v += __shfl_down(v, off, 32);
    float inv = rsqrtf(__shfl(v, 0, 32) / (float)D + 1e-5f);
    float rwv = 1.0f / (1.0f + __expf(-rw[0]));
    const float* pr = proj + (size_t)row * D;
    float* o = out + (size_t)row * D;
    for (int c = lane; c < D; c += 32) {
        float val = (xr[c] - mu) * inv * g[c] + b[c] + rwv * pr[c];
        if (do_elu) val = (val > 0.0f) ? val : (__expf(val) - 1.0f);
        o[c] = val;
    }
}

// scores[n] = <p1[n,:32], w2> + b2.  One wave per node.
__global__ __launch_bounds__(256) void pool_score_kernel(
    const float* __restrict__ p1, const float* __restrict__ w2,
    const float* __restrict__ b2, float* __restrict__ scores, int n)
{
    int row  = (blockIdx.x * blockDim.x + threadIdx.x) >> 5;
    int lane = threadIdx.x & 31;
    if (row >= n) return;
    float v = p1[(size_t)row * 32 + lane] * w2[lane];
    for (int off = 16; off > 0; off >>= 1) v += __shfl_down(v, off, 32);
    if (lane == 0) scores[row] = v + b2[0];
}

// softmax over N scores + weighted sum g[64] = sum_n f2[n,:]*w[n]. One block.
__global__ __launch_bounds__(1024) void pool_softmax_kernel(
    const float* __restrict__ f2, const float* __restrict__ scores,
    float* __restrict__ gout, int n)
{
    __shared__ float red[1024];
    int tid = threadIdx.x;
    float m = -3.4e38f;
    for (int i = tid; i < n; i += 1024) m = fmaxf(m, scores[i]);
    red[tid] = m; __syncthreads();
    for (int off = 512; off > 0; off >>= 1) {
        if (tid < off) red[tid] = fmaxf(red[tid], red[tid + off]);
        __syncthreads();
    }
    float mx = red[0]; __syncthreads();
    float s = 0.0f;
    for (int i = tid; i < n; i += 1024) s += __expf(scores[i] - mx);
    red[tid] = s; __syncthreads();
    for (int off = 512; off > 0; off >>= 1) {
        if (tid < off) red[tid] += red[tid + off];
        __syncthreads();
    }
    float sum = red[0]; __syncthreads();
    int c = tid & 63, k = tid >> 6;                        // 16 node groups
    float acc = 0.0f;
    for (int i = k; i < n; i += 16)
        acc += f2[(size_t)i * 64 + c] * __expf(scores[i] - mx);
    red[tid] = acc; __syncthreads();
    if (tid < 64) {
        float a = 0.0f;
        for (int kk = 0; kk < 16; ++kk) a += red[tid + (kk << 6)];
        gout[tid] = a / sum;
    }
}

// cat[n,192] = [f2 | f3 | g]
__global__ void build_cat_kernel(const float* __restrict__ f2,
                                 const float* __restrict__ f3,
                                 const float* __restrict__ g,
                                 float* __restrict__ cat, int n)
{
    int idx = blockIdx.x * blockDim.x + threadIdx.x;
    if (idx >= n * 192) return;
    int j = idx % 192, r = idx / 192;
    float v = (j < 64)  ? f2[(size_t)r * 64 + j]
            : (j < 128) ? f3[(size_t)r * 64 + (j - 64)]
                        : g[j - 128];
    cat[idx] = v;
}

// ---------------------------------------------------------------------------
// Host driver
// ---------------------------------------------------------------------------
struct LayerP {
    const float *W, *att_dst, *att_src, *bias, *ln_b, *ln_g, *resW, *res_b, *rw;
};

static inline int ceil_div_i(int a, int b) { return (a + b - 1) / b; }

extern "C" void kernel_launch(void* const* d_in, const int* in_sizes, int n_in,
                              void* d_out, int out_size, void* d_ws, size_t ws_size,
                              hipStream_t stream)
{
    (void)n_in; (void)out_size; (void)ws_size;
    const int NN = NNODES;

    // ---- input unpack: top-level insertion order; params as JAX pytree
    //      (dict keys sorted; None leaves dropped) --------------------------
    int ii = 0;
    const float* xf = (const float*)d_in[ii++];                       // x [6000,128]
    auto read_layer = [&](bool has_res) {
        LayerP L;
        L.W       = (const float*)d_in[ii++];
        L.att_dst = (const float*)d_in[ii++];
        L.att_src = (const float*)d_in[ii++];
        L.bias    = (const float*)d_in[ii++];
        L.ln_b    = (const float*)d_in[ii++];
        L.ln_g    = (const float*)d_in[ii++];
        if (has_res) { L.resW = (const float*)d_in[ii++]; L.res_b = (const float*)d_in[ii++]; }
        else         { L.resW = nullptr;                  L.res_b = nullptr; }
        L.rw      = (const float*)d_in[ii++];
        return L;
    };
    LayerP enc0[2], enc1[3];
    enc0[0] = read_layer(true);  enc0[1] = read_layer(true);
    enc1[0] = read_layer(true);  enc1[1] = read_layer(false); enc1[2] = read_layer(true);
    const float* fus_b1 = (const float*)d_in[ii++];
    const float* fus_b2 = (const float*)d_in[ii++];
    const float* fus_w1 = (const float*)d_in[ii++];
    const float* fus_w2 = (const float*)d_in[ii++];
    const float* pool_b1 = (const float*)d_in[ii++];
    const float* pool_b2 = (const float*)d_in[ii++];
    const float* pool_w1 = (const float*)d_in[ii++];
    const float* pool_w2 = (const float*)d_in[ii++];
    ii++;                                                    // edge_index (unused)
    const int* ek2 = (const int*)d_in[ii]; int E2 = in_sizes[ii] / 2; ii++;
    const int* ek3 = (const int*)d_in[ii]; int E3 = in_sizes[ii] / 2; ii++;
    const int *k2s = ek2, *k2d = ek2 + E2;
    const int *k3s = ek3, *k3d = ek3 + E3;

    // ---- workspace layout (fp32 elements) --------------------------------
    float* ws = (float*)d_ws;
    size_t off = 0;
    auto alloc = [&](size_t cnt) { float* p = ws + off; off += cnt; return p; };
    float*    hbuf  = alloc((size_t)NN * 256);
    float*    agg   = alloc((size_t)NN * 256);
    float*    proj  = alloc((size_t)NN * 256);
    float*    xa    = alloc((size_t)NN * 256);
    float*    xb    = alloc((size_t)NN * 256);
    float*    asrc  = alloc((size_t)NN * 4);
    float*    adst  = alloc((size_t)NN * 4);
    float*    denom = alloc((size_t)NN * 4);
    unsigned* emax  = (unsigned*)alloc((size_t)NN * 4);
    float*    f2    = alloc((size_t)NN * 64);
    float*    f3    = alloc((size_t)NN * 64);
    float*    p1    = alloc((size_t)NN * 32);
    float*    scr   = alloc((size_t)NN);
    float*    gvec  = alloc(64);
    float*    cat   = alloc((size_t)NN * 192);
    float*    hid   = alloc((size_t)NN * 128);
    int Emaxc = (E2 > E3 ? E2 : E3) + NN;
    float*    eexp  = alloc((size_t)Emaxc * 4);

    auto gemm = [&](const float* A, const float* B, const float* bias, float* C,
                    int M, int K, int Ncol, int act) {
        dim3 grid(ceil_div_i(M, 64), ceil_div_i(Ncol, 64));
        size_t shbytes = (size_t)256 * K;          // A panel (128*K B) + B panel
        wmma_gemm_kernel<<<grid, 128, shbytes, stream>>>(A, B, bias, C, M, K, Ncol, act);
    };

    auto run_enc = [&](const int* es, const int* ed, int E,
                       const LayerP* L, int nl, float* fout) {
        const float* xin = xf;
        int cur = 128;
        for (int i = 0; i < nl; ++i) {
            bool last = (i == nl - 1);
            int  Da   = last ? 64 : 256;
            // h = xin @ W  (WMMA)
            gemm(xin, L[i].W, nullptr, hbuf, NN, cur, 256, 0);
            attn_scores_kernel<<<ceil_div_i(NN * 32, 256), 256, 0, stream>>>(
                hbuf, L[i].att_src, L[i].att_dst, asrc, adst, NN);
            fill_u32_kernel<<<ceil_div_i(NN * 4, 256), 256, 0, stream>>>(emax, 0u, NN * 4);
            fill_f32_kernel<<<ceil_div_i(NN * 4, 256), 256, 0, stream>>>(denom, 0.0f, NN * 4);
            fill_f32_kernel<<<ceil_div_i(NN * 256, 256), 256, 0, stream>>>(agg, 0.0f, NN * 256);
            int Etot = E + NN;
            edge_max_kernel<<<ceil_div_i(Etot * 4, 256), 256, 0, stream>>>(
                es, ed, E, NN, asrc, adst, emax);
            edge_exp_kernel<<<ceil_div_i(Etot * 4, 256), 256, 0, stream>>>(
                es, ed, E, NN, asrc, adst, emax, eexp, denom);
            long long tot = (long long)Etot * 256;
            aggregate_kernel<<<(unsigned)((tot + 255) / 256), 256, 0, stream>>>(
                es, ed, E, NN, hbuf, eexp, denom, agg);
            // heads combine + bias -> reuse hbuf as xg (aggregate already done)
            if (!last)
                bias_concat_kernel<<<ceil_div_i(NN * 256, 256), 256, 0, stream>>>(
                    agg, L[i].bias, hbuf, NN);
            else
                bias_mean_kernel<<<ceil_div_i(NN * 64, 256), 256, 0, stream>>>(
                    agg, L[i].bias, hbuf, NN);
            // residual projection (WMMA) or identity
            const float* pp;
            if (L[i].resW) {
                gemm(xin, L[i].resW, L[i].res_b, proj, NN, cur, Da, 0);
                pp = proj;
            } else {
                pp = xin;   // cur == Da == 256 in this case
            }
            float* xout = last ? fout : ((i & 1) ? xb : xa);
            ln_res_kernel<<<ceil_div_i(NN * 32, 256), 256, 0, stream>>>(
                hbuf, L[i].ln_g, L[i].ln_b, pp, L[i].rw, xout, NN, Da, last ? 0 : 1);
            xin = xout;
            cur = Da;
        }
    };

    // ---- encoders --------------------------------------------------------
    run_enc(k2s, k2d, E2, enc0, 2, f2);   // scale-2 encoder -> f2 [N,64]
    run_enc(k3s, k3d, E3, enc1, 3, f3);   // scale-3 encoder -> f3 [N,64]

    // ---- attention pooling ----------------------------------------------
    gemm(f2, pool_w1, pool_b1, p1, NN, 64, 32, 2 /*tanh*/);
    pool_score_kernel<<<ceil_div_i(NN * 32, 256), 256, 0, stream>>>(
        p1, pool_w2, pool_b2, scr, NN);
    pool_softmax_kernel<<<1, 1024, 0, stream>>>(f2, scr, gvec, NN);

    // ---- fusion ----------------------------------------------------------
    build_cat_kernel<<<ceil_div_i(NN * 192, 256), 256, 0, stream>>>(f2, f3, gvec, cat, NN);
    gemm(cat, fus_w1, fus_b1, hid, NN, 192, 128, 1 /*relu*/);
    gemm(hid, fus_w2, fus_b2, (float*)d_out, NN, 128, 64, 0);
}
